// GPT_35158602285630
// MI455X (gfx1250) — matrix operand order repaired and checked by reading, not compiled, not run on hardware
//
#include <hip/hip_runtime.h>
#include <hip/hip_bf16.h>

// ---------------------------------------------------------------------------
// CDNA5 (gfx1250) causal attention block: cast -> QKV proj -> flash attn -> out proj
// Matmuls via v_wmma_f32_16x16x32_bf16 (wave32); GEMM staging via async
// global->LDS DMA (ASYNCcnt) with double buffering.
// ---------------------------------------------------------------------------

typedef __attribute__((ext_vector_type(16))) __bf16 v16bf;
typedef __attribute__((ext_vector_type(8)))  float  v8f;
typedef int v4i __attribute__((vector_size(16)));
typedef __attribute__((address_space(1))) v4i* gptr_v4i;   // global (AS1) v4i*
typedef __attribute__((address_space(3))) v4i* lptr_v4i;   // LDS (AS3) v4i*

#if __has_builtin(__builtin_amdgcn_global_load_async_to_lds_b128) && \
    __has_builtin(__builtin_amdgcn_s_wait_asynccnt)
#define HAS_ASYNC_LDS 1
#else
#define HAS_ASYNC_LDS 0
#endif

union BF16Frag {
    unsigned short u[16];
    uint4          q[2];
    v16bf          v;
};

__device__ __forceinline__ unsigned short f2bf(float f) {
    union { float f; unsigned u; } x;
    x.f = f;
    unsigned r = x.u + 0x7FFFu + ((x.u >> 16) & 1u);   // round-to-nearest-even
    return (unsigned short)(r >> 16);
}

__device__ __forceinline__ v8f zero8() {
    v8f z;
#pragma unroll
    for (int i = 0; i < 8; ++i) z[i] = 0.0f;
    return z;
}

// A-matrix fragment (16x32 bf16), ISA layout:
// lane L (row = L&15): halves 0..7 = K{kb..kb+7}, halves 8..15 = K{16+kb..+7},
// kb = (L>=16)*8.  rowptr points at the 32-wide K window for this row.
__device__ __forceinline__ v16bf load_frag_a(const unsigned short* rowptr, int lane) {
    const int kb = (lane >> 4) << 3;   // 0 or 8
    BF16Frag f;
    f.q[0] = *(const uint4*)(rowptr + kb);
    f.q[1] = *(const uint4*)(rowptr + 16 + kb);
    return f.v;
}

// B-matrix fragment (32x16 bf16): lane L (col = L&15): halves 0..7 = K{koff..koff+7},
// halves 8..15 = K{16+koff..+7}, koff = (L>=16)*8.  rowptr = 32-wide K window for this column (K contiguous).
__device__ __forceinline__ v16bf load_frag_b(const unsigned short* rowptr, int lane) {
    const int koff = (lane >> 4) << 3; // 0 or 8
    BF16Frag f;
    f.q[0] = *(const uint4*)(rowptr + koff);
    f.q[1] = *(const uint4*)(rowptr + 16 + koff);
    return f.v;
}

// ---------------------------------------------------------------------------
// One-pass fp32 -> bf16 cast (memory-bound; removes all cvt work from GEMMs).
// ---------------------------------------------------------------------------
__global__ __launch_bounds__(256) void cast_f32_bf16_kernel(
    const float* __restrict__ in, unsigned short* __restrict__ out, int n)
{
    const int i = (blockIdx.x * 256 + threadIdx.x) * 8;
    if (i >= n) return;
    float4 a = *(const float4*)(in + i);
    float4 b = *(const float4*)(in + i + 4);
    union { unsigned short u[8]; uint4 q; } o;
    o.u[0] = f2bf(a.x); o.u[1] = f2bf(a.y); o.u[2] = f2bf(a.z); o.u[3] = f2bf(a.w);
    o.u[4] = f2bf(b.x); o.u[5] = f2bf(b.y); o.u[6] = f2bf(b.z); o.u[7] = f2bf(b.w);
    *(uint4*)(out + i) = o.q;
}

// ---------------------------------------------------------------------------
// GEMM: C[M,N] = A[M,K] * B[N,K]^T, A/B bf16, C fp32 or bf16.
// Block tile 128(M) x 256(N), BK=32 per stage, 256 threads = 8 waves.
// Wave tile 64x64 (16 WMMAs per k-step, 16 ds_b128 -> 1.0 loads/WMMA).
// Double-buffered async global->LDS staging.
// ---------------------------------------------------------------------------
#define GK 32
#define GPAD 8

template <bool OUT_F32>
__global__ __launch_bounds__(256) void gemm_bt_kernel(
    const unsigned short* __restrict__ Ap, const unsigned short* __restrict__ Bp,
    void* __restrict__ Cp, int M, int N, int K)
{
    __shared__ unsigned short As[2][128][GK + GPAD];   // 20 KB
    __shared__ unsigned short Bs[2][256][GK + GPAD];   // 40 KB

    const int tid  = threadIdx.x;
    const int lane = tid & 31;
    const int wid  = tid >> 5;
    const int wm   = (wid >> 2) * 64;   // 0,64
    const int wn   = (wid & 3) * 64;    // 0,64,128,192
    const int m0   = blockIdx.y * 128;
    const int n0   = blockIdx.x * 256;
    const int ln   = lane & 15;
    const int mhi  = (lane >> 4) << 3;

    // staging: A -> (row=tid>>1, 16-wide half), B -> (row=tid, full 32)
    const int lrA = tid >> 1;
    const int lcA = (tid & 1) * 16;

    const unsigned short* Asrc = Ap + (size_t)(m0 + lrA) * K + lcA;
    const unsigned short* Bsrc = Bp + (size_t)(n0 + tid) * K;

    v8f acc[4][4];
#pragma unroll
    for (int i = 0; i < 4; ++i)
#pragma unroll
        for (int j = 0; j < 4; ++j) acc[i][j] = zero8();

    const int nk = K / GK;

#if HAS_ASYNC_LDS
    // ---- async double-buffered staging (6 async b128 loads per thread/tile) ----
    {
        __builtin_amdgcn_global_load_async_to_lds_b128(
            (gptr_v4i)(Asrc), (lptr_v4i)&As[0][lrA][lcA], 0, 0);
        __builtin_amdgcn_global_load_async_to_lds_b128(
            (gptr_v4i)(Asrc + 8), (lptr_v4i)&As[0][lrA][lcA + 8], 0, 0);
#pragma unroll
        for (int i = 0; i < 4; ++i)
            __builtin_amdgcn_global_load_async_to_lds_b128(
                (gptr_v4i)(Bsrc + i * 8), (lptr_v4i)&Bs[0][tid][i * 8], 0, 0);
    }
    for (int ki = 0; ki < nk; ++ki) {
        const int buf = ki & 1;
        if (ki + 1 < nk) {
            const int kn = (ki + 1) * GK;
            __builtin_amdgcn_global_load_async_to_lds_b128(
                (gptr_v4i)(Asrc + kn), (lptr_v4i)&As[buf ^ 1][lrA][lcA], 0, 0);
            __builtin_amdgcn_global_load_async_to_lds_b128(
                (gptr_v4i)(Asrc + kn + 8), (lptr_v4i)&As[buf ^ 1][lrA][lcA + 8], 0, 0);
#pragma unroll
            for (int i = 0; i < 4; ++i)
                __builtin_amdgcn_global_load_async_to_lds_b128(
                    (gptr_v4i)(Bsrc + kn + i * 8),
                    (lptr_v4i)&Bs[buf ^ 1][tid][i * 8], 0, 0);
            __builtin_amdgcn_s_wait_asynccnt(6);  // current tile resident
        } else {
            __builtin_amdgcn_s_wait_asynccnt(0);
        }
        __syncthreads();

        v16bf af[4], bf[4];
#pragma unroll
        for (int tm = 0; tm < 4; ++tm)
            af[tm] = load_frag_a(&As[buf][wm + tm * 16 + ln][0], lane);
#pragma unroll
        for (int tn = 0; tn < 4; ++tn)
            bf[tn] = load_frag_b(&Bs[buf][wn + tn * 16 + ln][0], lane);
#pragma unroll
        for (int tm = 0; tm < 4; ++tm)
#pragma unroll
            for (int tn = 0; tn < 4; ++tn)
                acc[tm][tn] = __builtin_amdgcn_wmma_f32_16x16x32_bf16(
                    false, af[tm], false, bf[tn], (short)0, acc[tm][tn], false, false);
        __syncthreads();   // protect buf before it is refilled next iteration
    }
#else
    // ---- fallback: synchronous staging ----
    for (int ki = 0; ki < nk; ++ki) {
        const int k0 = ki * GK;
        __syncthreads();
        *(uint4*)&As[0][lrA][lcA]     = *(const uint4*)(Asrc + k0);
        *(uint4*)&As[0][lrA][lcA + 8] = *(const uint4*)(Asrc + k0 + 8);
#pragma unroll
        for (int i = 0; i < 4; ++i)
            *(uint4*)&Bs[0][tid][i * 8] = *(const uint4*)(Bsrc + k0 + i * 8);
        if (ki + 1 < nk) {
            __builtin_prefetch(Asrc + k0 + GK, 0, 0);
            __builtin_prefetch(Bsrc + k0 + GK, 0, 0);
        }
        __syncthreads();

        v16bf af[4], bf[4];
#pragma unroll
        for (int tm = 0; tm < 4; ++tm)
            af[tm] = load_frag_a(&As[0][wm + tm * 16 + ln][0], lane);
#pragma unroll
        for (int tn = 0; tn < 4; ++tn)
            bf[tn] = load_frag_b(&Bs[0][wn + tn * 16 + ln][0], lane);
#pragma unroll
        for (int tm = 0; tm < 4; ++tm)
#pragma unroll
            for (int tn = 0; tn < 4; ++tn)
                acc[tm][tn] = __builtin_amdgcn_wmma_f32_16x16x32_bf16(
                    false, af[tm], false, bf[tn], (short)0, acc[tm][tn], false, false);
    }
#endif

    // epilogue: C/D layout — vgpr r, lanes 0-15 -> M=r, lanes 16-31 -> M=r+8
#pragma unroll
    for (int tm = 0; tm < 4; ++tm)
#pragma unroll
        for (int tn = 0; tn < 4; ++tn)
#pragma unroll
            for (int r = 0; r < 8; ++r) {
                const int m = m0 + wm + tm * 16 + r + mhi;
                const int n = n0 + wn + tn * 16 + ln;
                const float vv = acc[tm][tn][r];
                if (OUT_F32)
                    ((float*)Cp)[(size_t)m * N + n] = vv;
                else
                    ((unsigned short*)Cp)[(size_t)m * N + n] = f2bf(vv);
            }
}

// ---------------------------------------------------------------------------
// Flash attention (causal), bf16 Q/K/V [B*T, H*128] row-major, bf16 out Y.
// Block = 128 threads = 4 waves; block covers 64 query rows of one (b,h);
// wave w owns 16 rows.  Key blocks of 32, uniform trip count per block.
// ---------------------------------------------------------------------------
#define ATT_T   2048
#define ATT_H   16
#define ATT_D   128
#define ATT_C   (ATT_H * ATT_D)

__global__ __launch_bounds__(128) void attn_kernel(
    const unsigned short* __restrict__ Q,
    const unsigned short* __restrict__ Km,
    const unsigned short* __restrict__ Vm,
    unsigned short* __restrict__ Y)
{
    __shared__ unsigned short Ks[32][ATT_D + 8];   // [key][d]
    __shared__ unsigned short Vt[ATT_D][32 + 8];   // [d][key]  (transposed)
    __shared__ unsigned short Ps[4][16][32];       // per-wave P tile

    const int tid  = threadIdx.x;
    const int lane = tid & 31;
    const int wid  = tid >> 5;
    const int bh   = blockIdx.x;
    const int b    = bh >> 4;
    const int h    = bh & 15;
    const int qb   = blockIdx.y;          // 64-row block index
    const int qrow0 = qb * 64 + wid * 16; // wave's first query row
    const int col0  = h * ATT_D;
    const int ln    = lane & 15;
    const int kb    = (lane >> 4) << 3;   // A-frag K base (0/8)
    const int mhi   = kb;                 // C-layout row offset (0/8)

    // resident Q fragments: 4 K-steps of 32 over d=0..127
    v16bf qf[4];
    {
        const unsigned short* qrow =
            Q + (size_t)(b * ATT_T + qrow0 + ln) * ATT_C + col0;
#pragma unroll
        for (int s = 0; s < 4; ++s) {
            BF16Frag f;
            f.q[0] = *(const uint4*)(qrow + s * 32 + kb);
            f.q[1] = *(const uint4*)(qrow + s * 32 + 16 + kb);
            qf[s] = f.v;
        }
    }

    v8f acc[8];
    float mstat[8], lstat[8];
#pragma unroll
    for (int r = 0; r < 8; ++r) {
        acc[r] = zero8();
        mstat[r] = -3.0e38f;
        lstat[r] = 0.0f;
    }

    const float scale = 0.08838834764831845f;  // 1/sqrt(128)
    const int jmax = qb * 2 + 2;               // uniform across the block

    for (int j = 0; j < jmax; ++j) {
        const int kbase = j * 32;
        __syncthreads();
        // ---- cooperatively stage K (row-major) and V (transposed) ----
        {
            const int krow = tid >> 2;       // 0..31
            const int d0   = (tid & 3) * 32; // 0,32,64,96
            const unsigned short* ksrc =
                Km + (size_t)(b * ATT_T + kbase + krow) * ATT_C + col0 + d0;
#pragma unroll
            for (int i = 0; i < 4; ++i)
                *(uint4*)&Ks[krow][d0 + i * 8] = *(const uint4*)(ksrc + i * 8);

            const unsigned short* vsrc =
                Vm + (size_t)(b * ATT_T + kbase + krow) * ATT_C + col0 + d0;
            union { uint4 q[4]; unsigned short u[32]; } tv;
#pragma unroll
            for (int i = 0; i < 4; ++i) tv.q[i] = *(const uint4*)(vsrc + i * 8);
#pragma unroll
            for (int i = 0; i < 32; ++i) Vt[d0 + i][krow] = tv.u[i];
        }
        __syncthreads();

        // ---- S = Q K^T : two 16-key tiles ----
        v8f st[2];
#pragma unroll
        for (int hh = 0; hh < 2; ++hh) {
            v8f s = zero8();
            const unsigned short* krow = &Ks[hh * 16 + ln][0];
#pragma unroll
            for (int s4 = 0; s4 < 4; ++s4) {
                v16bf kf = load_frag_b(krow + s4 * 32, lane);
                s = __builtin_amdgcn_wmma_f32_16x16x32_bf16(
                    false, qf[s4], false, kf, (short)0, s, false, false);
            }
            st[hh] = s;
        }

        // ---- online softmax (per row; each 16-lane group holds one row set) ----
#pragma unroll
        for (int r = 0; r < 8; ++r) {
            const int qg  = qrow0 + r + mhi;      // global query row
            const int k0i = kbase + ln;
            const int k1i = kbase + 16 + ln;
            float v0 = (k0i <= qg) ? st[0][r] * scale : -3.0e38f;
            float v1 = (k1i <= qg) ? st[1][r] * scale : -3.0e38f;
            float rm = fmaxf(v0, v1);
#pragma unroll
            for (int off = 1; off < 16; off <<= 1)
                rm = fmaxf(rm, __shfl_xor(rm, off, 32));
            const float mnew  = fmaxf(mstat[r], rm);
            const float alpha = __expf(mstat[r] - mnew);
            const float p0 = __expf(v0 - mnew);
            const float p1 = __expf(v1 - mnew);
            float ps = p0 + p1;
#pragma unroll
            for (int off = 1; off < 16; off <<= 1)
                ps += __shfl_xor(ps, off, 32);
            lstat[r] = lstat[r] * alpha + ps;
            mstat[r] = mnew;
#pragma unroll
            for (int t = 0; t < 8; ++t) acc[t][r] *= alpha;

            const int prow = r + mhi;
            Ps[wid][prow][ln]      = f2bf(p0);
            Ps[wid][prow][16 + ln] = f2bf(p1);
        }
        __syncthreads();

        // ---- O += P (16x32) x V (32x128) ----
        v16bf pf = load_frag_a(&Ps[wid][ln][0], lane);
#pragma unroll
        for (int t = 0; t < 8; ++t) {
            v16bf vf = load_frag_b(&Vt[t * 16 + ln][0], lane);
            acc[t] = __builtin_amdgcn_wmma_f32_16x16x32_bf16(
                false, pf, false, vf, (short)0, acc[t], false, false);
        }
    }

    // ---- normalize and write Y (bf16) ----
#pragma unroll
    for (int t = 0; t < 8; ++t)
#pragma unroll
        for (int r = 0; r < 8; ++r) {
            const int m = qrow0 + r + mhi;
            const float y = acc[t][r] / lstat[r];
            Y[(size_t)(b * ATT_T + m) * ATT_C + col0 + t * 16 + ln] = f2bf(y);
        }
}

// ---------------------------------------------------------------------------
// Launch: x[B,T,C] f32, Wq/Wk/Wv [2048,2048] f32, Wo [2048,2048] f32.
// out = attention_block(x) f32.
// Workspace (bf16): xb[M*C], wq/wk/wv/wo[C*C], Q/K/V/Y[M*C]  (~201 MB).
// ---------------------------------------------------------------------------
extern "C" void kernel_launch(void* const* d_in, const int* in_sizes, int n_in,
                              void* d_out, int out_size, void* d_ws, size_t ws_size,
                              hipStream_t stream)
{
    (void)in_sizes; (void)n_in; (void)out_size; (void)ws_size;

    const float* x  = (const float*)d_in[0];
    const float* Wq = (const float*)d_in[1];
    const float* Wk = (const float*)d_in[2];
    const float* Wv = (const float*)d_in[3];
    const float* Wo = (const float*)d_in[4];
    float* out = (float*)d_out;

    const int Bz = 4, T = 2048, C = 2048;
    const int M  = Bz * T;                 // 8192
    const size_t MC = (size_t)M * C;
    const size_t CC = (size_t)C * C;

    unsigned short* xb  = (unsigned short*)d_ws;
    unsigned short* wqb = xb  + MC;
    unsigned short* wkb = wqb + CC;
    unsigned short* wvb = wkb + CC;
    unsigned short* wob = wvb + CC;
    unsigned short* Qb  = wob + CC;
    unsigned short* Kb  = Qb  + MC;
    unsigned short* Vb  = Kb  + MC;
    unsigned short* Yb  = Vb  + MC;

    // ---- one-pass casts (memory-bound) ----
    cast_f32_bf16_kernel<<<(int)(MC / 8 / 256), 256, 0, stream>>>(x,  xb,  (int)MC);
    cast_f32_bf16_kernel<<<(int)(CC / 8 / 256), 256, 0, stream>>>(Wq, wqb, (int)CC);
    cast_f32_bf16_kernel<<<(int)(CC / 8 / 256), 256, 0, stream>>>(Wk, wkb, (int)CC);
    cast_f32_bf16_kernel<<<(int)(CC / 8 / 256), 256, 0, stream>>>(Wv, wvb, (int)CC);
    cast_f32_bf16_kernel<<<(int)(CC / 8 / 256), 256, 0, stream>>>(Wo, wob, (int)CC);

    dim3 gblk(C / 256, M / 128);           // (8, 64)

    gemm_bt_kernel<false><<<gblk, 256, 0, stream>>>(xb, wqb, Qb, M, C, C);
    gemm_bt_kernel<false><<<gblk, 256, 0, stream>>>(xb, wkb, Kb, M, C, C);
    gemm_bt_kernel<false><<<gblk, 256, 0, stream>>>(xb, wvb, Vb, M, C, C);

    attn_kernel<<<dim3(Bz * ATT_H, T / 64), 128, 0, stream>>>(Qb, Kb, Vb, Yb);

    gemm_bt_kernel<true><<<gblk, 256, 0, stream>>>(Yb, wob, out, M, C, C);
}
